// SetAbstraction_63857573756989
// MI455X (gfx1250) — matrix operand order repaired and checked by reading, not compiled, not run on hardware
//
#include <hip/hip_runtime.h>

#define NPTS  65536
#define MCENT 4096
#define KNN   64
#define R2    0.04f   // RADIUS^2 = 0.2^2

typedef __attribute__((ext_vector_type(16))) _Float16 v16h;
typedef __attribute__((ext_vector_type(8)))  _Float16 v8h;
typedef __attribute__((ext_vector_type(8)))  float    v8f;

// ---------------------------------------------------------------------------
// Kernel 0: convert weights to f16, transposed ([n][k]) + zero-padded layouts
// suitable for contiguous per-lane WMMA B-fragment loads.
//   w1t: [64 n][32 k]  (k>=6 zero)
//   w2t: [64 n][64 k]
//   w3t: [128 n][64 k]
//   bias: b1(64) | b2(64) | b3(128) as f32
// ---------------------------------------------------------------------------
__global__ __launch_bounds__(256) void sa_prep_weights(
    const float* __restrict__ W1, const float* __restrict__ b1,
    const float* __restrict__ W2, const float* __restrict__ b2,
    const float* __restrict__ W3, const float* __restrict__ b3,
    _Float16* __restrict__ w1t, _Float16* __restrict__ w2t,
    _Float16* __restrict__ w3t, float* __restrict__ bias) {
  int t = blockIdx.x * blockDim.x + threadIdx.x;
  if (t < 64 * 32) {            // W1 [6,64] -> [64][32]
    int n = t >> 5, k = t & 31;
    w1t[t] = (_Float16)((k < 6) ? W1[k * 64 + n] : 0.0f);
  }
  if (t < 64 * 64) {            // W2 [64,64] -> [64][64]
    int n = t >> 6, k = t & 63;
    w2t[t] = (_Float16)W2[k * 64 + n];
  }
  if (t < 128 * 64) {           // W3 [64,128] -> [128][64]
    int n = t >> 6, k = t & 63;
    w3t[t] = (_Float16)W3[k * 128 + n];
  }
  if (t < 64)  bias[t]       = b1[t];
  if (t < 64)  bias[64 + t]  = b2[t];
  if (t < 128) bias[128 + t] = b3[t];
}

// ---------------------------------------------------------------------------
// Kernel 1: per-centroid exact K-th smallest squared distance via 4x8-bit
// radix select on the f32 bit pattern (monotonic for non-negative floats).
// CDNA5's 320KB LDS lets us keep all N=65536 d2 values (256KB) resident:
// distances are computed from global ONCE (fused with the round-0 histogram);
// the remaining 3 radix rounds and the gather pass scan LDS with b128 loads.
// One workgroup (256 threads) per centroid.
// ---------------------------------------------------------------------------
__global__ __launch_bounds__(256) void sa_knn_select(
    const float* __restrict__ verts, const int* __restrict__ cidx,
    int* __restrict__ nbr, int* __restrict__ cnt) {
  extern __shared__ __align__(16) float sd2[];   // NPTS floats = 256 KB
  __shared__ unsigned hist[256];
  __shared__ unsigned s_prefix, s_mask, s_kneed, s_cnt;
  __shared__ float s_c[3];

  const int m = blockIdx.x;
  const int t = threadIdx.x;

  if (t == 0) {
    int ci = cidx[m];
    s_c[0] = verts[3 * ci + 0];
    s_c[1] = verts[3 * ci + 1];
    s_c[2] = verts[3 * ci + 2];
    s_prefix = 0u; s_mask = 0u; s_kneed = KNN; s_cnt = 0u;
  }
  hist[t] = 0u;
  __syncthreads();
  const float cx = s_c[0], cy = s_c[1], cz = s_c[2];

  // ---- pass 0: compute d2 (4 points / iter, 3x float4 loads), store to LDS,
  //      and histogram the top byte in the same pass ----
  for (int j4 = t; j4 < NPTS / 4; j4 += 256) {
    const float4* v4 = (const float4*)(verts + 12 * j4);
    float4 a = v4[0], b = v4[1], c = v4[2];
    float dx, dy, dz;
    dx = a.x - cx; dy = a.y - cy; dz = a.z - cz;
    float d0 = fmaf(dx, dx, fmaf(dy, dy, dz * dz));
    dx = a.w - cx; dy = b.x - cy; dz = b.y - cz;
    float d1 = fmaf(dx, dx, fmaf(dy, dy, dz * dz));
    dx = b.z - cx; dy = b.w - cy; dz = c.x - cz;
    float d2v = fmaf(dx, dx, fmaf(dy, dy, dz * dz));
    dx = c.y - cx; dy = c.z - cy; dz = c.w - cz;
    float d3 = fmaf(dx, dx, fmaf(dy, dy, dz * dz));
    ((float4*)sd2)[j4] = make_float4(d0, d1, d2v, d3);
    atomicAdd(&hist[__float_as_uint(d0) >> 24], 1u);
    atomicAdd(&hist[__float_as_uint(d1) >> 24], 1u);
    atomicAdd(&hist[__float_as_uint(d2v) >> 24], 1u);
    atomicAdd(&hist[__float_as_uint(d3) >> 24], 1u);
  }
  __syncthreads();
  if (t == 0) {   // select top byte
    unsigned cum = 0, need = s_kneed;
    int b = 0;
    for (; b < 256; ++b) {
      unsigned h = hist[b];
      if (cum + h >= need) break;
      cum += h;
    }
    s_prefix = (unsigned)b << 24;
    s_mask   = 0xFFu << 24;
    s_kneed  = need - cum;
  }
  __syncthreads();

  // ---- rounds 1..3: refine from LDS-resident d2 ----
  for (int round = 1; round < 4; ++round) {
    const int shift = 24 - 8 * round;
    hist[t] = 0u;
    __syncthreads();
    const unsigned pmask = s_mask, pref = s_prefix;
    for (int j4 = t; j4 < NPTS / 4; j4 += 256) {
      float4 v = ((const float4*)sd2)[j4];
      unsigned u0 = __float_as_uint(v.x), u1 = __float_as_uint(v.y);
      unsigned u2 = __float_as_uint(v.z), u3 = __float_as_uint(v.w);
      if ((u0 & pmask) == pref) atomicAdd(&hist[(u0 >> shift) & 255u], 1u);
      if ((u1 & pmask) == pref) atomicAdd(&hist[(u1 >> shift) & 255u], 1u);
      if ((u2 & pmask) == pref) atomicAdd(&hist[(u2 >> shift) & 255u], 1u);
      if ((u3 & pmask) == pref) atomicAdd(&hist[(u3 >> shift) & 255u], 1u);
    }
    __syncthreads();
    if (t == 0) {
      unsigned cum = 0, need = s_kneed;
      int b = 0;
      for (; b < 256; ++b) {
        unsigned h = hist[b];
        if (cum + h >= need) break;
        cum += h;
      }
      s_prefix = s_prefix | ((unsigned)b << shift);
      s_mask   = pmask | (0xFFu << shift);
      s_kneed  = need - cum;
    }
    __syncthreads();
  }

  // ---- gather: indices with d2 <= min(kth d2, RADIUS^2), from LDS ----
  const float tau2 = fminf(__uint_as_float(s_prefix), R2);
  for (int j4 = t; j4 < NPTS / 4; j4 += 256) {
    float4 v = ((const float4*)sd2)[j4];
    int j = 4 * j4;
    if (v.x <= tau2) { unsigned p = atomicAdd(&s_cnt, 1u); if (p < KNN) nbr[m * KNN + p] = j; }
    if (v.y <= tau2) { unsigned p = atomicAdd(&s_cnt, 1u); if (p < KNN) nbr[m * KNN + p] = j + 1; }
    if (v.z <= tau2) { unsigned p = atomicAdd(&s_cnt, 1u); if (p < KNN) nbr[m * KNN + p] = j + 2; }
    if (v.w <= tau2) { unsigned p = atomicAdd(&s_cnt, 1u); if (p < KNN) nbr[m * KNN + p] = j + 3; }
  }
  __syncthreads();
  if (t == 0) cnt[m] = (int)min(s_cnt, (unsigned)KNN);
}

// ---------------------------------------------------------------------------
// Kernel 2: fused MLP (6->64->64->128, ReLU) via v_wmma_f32_16x16x32_f16 and
// masked max-aggregation. One workgroup (4 waves / 128 threads) per centroid;
// each wave owns a 16-row tile of the 64 neighbor rows.
// ---------------------------------------------------------------------------
__global__ __launch_bounds__(128) void sa_mlp_max(
    const float* __restrict__ verts, const int* __restrict__ cidx,
    const int* __restrict__ nbr, const int* __restrict__ cnt,
    const _Float16* __restrict__ w1t, const _Float16* __restrict__ w2t,
    const _Float16* __restrict__ w3t, const float* __restrict__ bias,
    float* __restrict__ out) {
  __shared__ __align__(16) _Float16 sW1[64 * 32];
  __shared__ __align__(16) _Float16 sW2[64 * 64];
  __shared__ __align__(16) _Float16 sW3[128 * 64];
  __shared__ __align__(16) _Float16 sF [64 * 32];   // features, K-padded to 32
  __shared__ __align__(16) _Float16 sH1[64 * 64];
  __shared__ __align__(16) _Float16 sH2[64 * 64];
  __shared__ float sBias[256];
  __shared__ float sMaxW[4][128];
  __shared__ int   sIdx[64];
  __shared__ int   sCnt;
  __shared__ float sC[3];

  const int m = blockIdx.x;
  const int t = threadIdx.x;

  // --- stage weights/biases into LDS (16B vector copies) ---
  {
    uint4*       d1 = (uint4*)sW1; const uint4* s1 = (const uint4*)w1t;
    for (int i = t; i < 256;  i += 128) d1[i] = s1[i];
    uint4*       d2 = (uint4*)sW2; const uint4* s2 = (const uint4*)w2t;
    for (int i = t; i < 512;  i += 128) d2[i] = s2[i];
    uint4*       d3 = (uint4*)sW3; const uint4* s3 = (const uint4*)w3t;
    for (int i = t; i < 1024; i += 128) d3[i] = s3[i];
    for (int i = t; i < 256;  i += 128) sBias[i] = bias[i];
  }
  if (t == 0) {
    sCnt = cnt[m];
    int ci = cidx[m];
    sC[0] = verts[3 * ci + 0];
    sC[1] = verts[3 * ci + 1];
    sC[2] = verts[3 * ci + 2];
  }
  if (t < 64) sIdx[t] = nbr[m * KNN + t];
  __syncthreads();

  // --- build feature rows: [pos_j, pos_j - pos_i], zero pad K to 32 ---
  if (t < 64) {
    bool valid = t < sCnt;
    int j = valid ? sIdx[t] : cidx[m];     // safe index for masked rows
    float px = verts[3 * j + 0], py = verts[3 * j + 1], pz = verts[3 * j + 2];
    _Float16* row = &sF[t * 32];
    row[0] = (_Float16)px;            row[1] = (_Float16)py;
    row[2] = (_Float16)pz;            row[3] = (_Float16)(px - sC[0]);
    row[4] = (_Float16)(py - sC[1]);  row[5] = (_Float16)(pz - sC[2]);
#pragma unroll
    for (int k = 6; k < 32; ++k) row[k] = (_Float16)0.0f;
  }
  __syncthreads();

  const int wave  = t >> 5;
  const int lane  = t & 31;
  const bool low  = lane < 16;
  const int rowA  = (wave << 4) + (lane & 15);   // A-fragment row for this lane
  const int koff  = low ? 0 : 8;                 // A K-offset per ISA layout
  const int kb16  = low ? 0 : 16;                // unused; loadB uses koff like A
  const int colN  = lane & 15;                   // B/C column for this lane
  const int rBase = (wave << 4) + (low ? 0 : 8); // C rows = rBase + vgpr

  // A fragment: row rowA, halves [ks+koff .. +7] and [ks+16+koff .. +7]
  auto loadA = [&](const _Float16* base, int stride, int ks) -> v16h {
    v8h lo = *(const v8h*)(base + rowA * stride + ks + koff);
    v8h hi = *(const v8h*)(base + rowA * stride + ks + 16 + koff);
    v16h a;
#pragma unroll
    for (int i = 0; i < 8; ++i) { a[i] = lo[i]; a[8 + i] = hi[i]; }
    return a;
  };
  // B fragment: column col, halves [ks+koff .. +7] and [ks+16+koff .. +7]
  auto loadB = [&](const _Float16* w, int stride, int col, int ks) -> v16h {
    const _Float16* p = w + col * stride + ks + koff;
    v8h lo = *(const v8h*)p;
    v8h hi = *(const v8h*)(p + 16);
    v16h b;
#pragma unroll
    for (int i = 0; i < 8; ++i) { b[i] = lo[i]; b[8 + i] = hi[i]; }
    return b;
  };

  // ---------------- Layer 1: [64x32] @ [32x64] ----------------
  {
    v16h A = loadA(sF, 32, 0);
    v8f acc[4];
#pragma unroll
    for (int c = 0; c < 4; ++c) {
      v16h B = loadB(sW1, 32, c * 16 + colN, 0);
      v8f z = {};
      acc[c] = __builtin_amdgcn_wmma_f32_16x16x32_f16(
          false, A, false, B, (short)0, z, false, false);
    }
#pragma unroll
    for (int c = 0; c < 4; ++c) {
      float bv = sBias[c * 16 + colN];
#pragma unroll
      for (int v = 0; v < 8; ++v) {
        float val = fmaxf(acc[c][v] + bv, 0.0f);
        sH1[(rBase + v) * 64 + c * 16 + colN] = (_Float16)val;
      }
    }
  }
  __syncthreads();

  // ---------------- Layer 2: [64x64] @ [64x64] ----------------
  {
    v8f acc[4] = {v8f{}, v8f{}, v8f{}, v8f{}};
#pragma unroll
    for (int s = 0; s < 2; ++s) {
      v16h A = loadA(sH1, 64, s * 32);
#pragma unroll
      for (int c = 0; c < 4; ++c) {
        v16h B = loadB(sW2, 64, c * 16 + colN, s * 32);
        acc[c] = __builtin_amdgcn_wmma_f32_16x16x32_f16(
            false, A, false, B, (short)0, acc[c], false, false);
      }
    }
#pragma unroll
    for (int c = 0; c < 4; ++c) {
      float bv = sBias[64 + c * 16 + colN];
#pragma unroll
      for (int v = 0; v < 8; ++v) {
        float val = fmaxf(acc[c][v] + bv, 0.0f);
        sH2[(rBase + v) * 64 + c * 16 + colN] = (_Float16)val;
      }
    }
  }
  __syncthreads();

  // ---------------- Layer 3: [64x64] @ [64x128] + masked row-max ----------
  {
    v8f acc[8] = {v8f{}, v8f{}, v8f{}, v8f{}, v8f{}, v8f{}, v8f{}, v8f{}};
#pragma unroll
    for (int s = 0; s < 2; ++s) {
      v16h A = loadA(sH2, 64, s * 32);
#pragma unroll
      for (int c = 0; c < 8; ++c) {
        v16h B = loadB(sW3, 64, c * 16 + colN, s * 32);
        acc[c] = __builtin_amdgcn_wmma_f32_16x16x32_f16(
            false, A, false, B, (short)0, acc[c], false, false);
      }
    }
    const int cN = sCnt;
#pragma unroll
    for (int c = 0; c < 8; ++c) {
      int colAbs = c * 16 + colN;
      float bv = sBias[128 + colAbs];
      float mx = -__builtin_inff();
#pragma unroll
      for (int v = 0; v < 8; ++v) {
        float val = acc[c][v] + bv;
        mx = (rBase + v < cN) ? fmaxf(mx, val) : mx;
      }
      float other = __shfl_xor(mx, 16, 32);   // fold rows held in upper lanes
      mx = fmaxf(mx, other);
      if (low) sMaxW[wave][colAbs] = mx;
    }
  }
  __syncthreads();

  // reduce 4 wave-tiles (4 x 16 rows) and write [128] output row
  {
    float a = fmaxf(sMaxW[0][t], sMaxW[1][t]);
    float b = fmaxf(sMaxW[2][t], sMaxW[3][t]);
    out[m * 128 + t] = fmaxf(a, b);
  }
}

// ---------------------------------------------------------------------------
extern "C" void kernel_launch(void* const* d_in, const int* in_sizes, int n_in,
                              void* d_out, int out_size, void* d_ws, size_t ws_size,
                              hipStream_t stream) {
  const float* verts = (const float*)d_in[0];
  const int*   cidx  = (const int*)d_in[1];
  const float* W1 = (const float*)d_in[2];
  const float* b1 = (const float*)d_in[3];
  const float* W2 = (const float*)d_in[4];
  const float* b2 = (const float*)d_in[5];
  const float* W3 = (const float*)d_in[6];
  const float* b3 = (const float*)d_in[7];
  float* out = (float*)d_out;

  char* ws = (char*)d_ws;
  _Float16* w1t  = (_Float16*)(ws + 0);                 // 4 KB
  _Float16* w2t  = (_Float16*)(ws + 4096);              // 8 KB
  _Float16* w3t  = (_Float16*)(ws + 12288);             // 16 KB
  float*    bias = (float*)   (ws + 28672);             // 1 KB
  int*      nbr  = (int*)     (ws + 29696);             // 1 MB
  int*      cnt  = (int*)     (ws + 29696 + MCENT * KNN * 4);  // 16 KB

  sa_prep_weights<<<32, 256, 0, stream>>>(W1, b1, W2, b2, W3, b3,
                                          w1t, w2t, w3t, bias);
  // 256 KB dynamic LDS: whole d2 array stays LDS-resident (CDNA5: 320 KB/WG)
  sa_knn_select<<<MCENT, 256, NPTS * sizeof(float), stream>>>(verts, cidx, nbr, cnt);
  sa_mlp_max<<<MCENT, 128, 0, stream>>>(verts, cidx, nbr, cnt,
                                        w1t, w2t, w3t, bias, out);
}
